// MultiHeadAttention_25675314496086
// MI455X (gfx1250) — matrix operand order repaired and checked by reading, not compiled, not run on hardware
//
#include <hip/hip_runtime.h>

// ---------------------------------------------------------------------------
// MI455X (gfx1250) multi-head attention, flash style, bf16 WMMA w/ f32 accum.
// B=2, L=2048, E=1024, H=16, D=64.
// Round 3: double-buffered global_load_async_to_lds_b128 prefetch (ASYNCcnt)
// overlapping WMMA; LDS-shared B panels in both GEMM kernels.
// ---------------------------------------------------------------------------

#define DEV static __device__ __forceinline__

typedef __bf16 bf16_t;
typedef __bf16 v16bf __attribute__((ext_vector_type(16)));
typedef float  v8f   __attribute__((ext_vector_type(8)));
typedef unsigned uq4 __attribute__((ext_vector_type(4)));

constexpr int Bn = 2, Ln = 2048, En = 1024, Hn = 16, Dn = 64;

// wave32 xor-swizzle (group-of-32 mode: offset = xor<<10 | or<<5 | and)
template <int PAT> DEV float swz(float v) {
  return __builtin_bit_cast(float,
      __builtin_amdgcn_ds_swizzle(__builtin_bit_cast(int, v), PAT));
}
// reduce across the 16 lanes of a half-wave (xor 1,2,4,8 stays inside half)
DEV float rowmax16(float v) {
  v = fmaxf(v, swz<0x041f>(v)); v = fmaxf(v, swz<0x081f>(v));
  v = fmaxf(v, swz<0x101f>(v)); v = fmaxf(v, swz<0x201f>(v));
  return v;
}
DEV float rowsum16(float v) {
  v += swz<0x041f>(v); v += swz<0x081f>(v);
  v += swz<0x101f>(v); v += swz<0x201f>(v);
  return v;
}

DEV v8f wmma_bf16(v16bf a, v16bf b, v8f c) {
  return __builtin_amdgcn_wmma_f32_16x16x32_bf16(
      false, a, false, b, (short)0, c, false, false);
}

// Load a 16-element bf16 tile fragment from two 16B-aligned 8-element runs.
struct Pair32 { uq4 a, b; };
DEV v16bf ldpair(const bf16_t* p0, const bf16_t* p1) {
  Pair32 s;
  s.a = *(const uq4*)p0;
  s.b = *(const uq4*)p1;
  return __builtin_bit_cast(v16bf, s);
}
// B-tile fragment: 16 contiguous bf16 (32B, 16B-aligned is enough).
DEV v16bf ld32(const bf16_t* p) { return ldpair(p, p + 8); }

// A-tile (16x32 bf16): lane -> row = lane&15; element e -> K = half*16+e
// i.e. two contiguous 8-elt runs at K = half*16 and K = half*16+8.
// B-tile (32x16 bf16): lane&15 -> N; element e -> K = half*16+e (contiguous).
// C/D f32 16x16: vgpr r -> row r+8*half, lane&15 -> col.

// LDS byte offset of a generic pointer into shared memory (aperture|offset).
DEV unsigned lds_off(const void* p) { return (unsigned)(uintptr_t)p; }

// One 16-byte async DMA: global -> LDS (ASYNCcnt-tracked, no VGPR data).
DEV void async_b128(unsigned lds_byte, const void* gaddr) {
  unsigned long long ga = (unsigned long long)(uintptr_t)gaddr;
  asm volatile("global_load_async_to_lds_b128 %0, %1, off"
               :: "v"(lds_byte), "v"(ga) : "memory");
}
DEV void wait_async0() { asm volatile("s_wait_asynccnt 0x0" ::: "memory"); }

// ---------------------------------------------------------------------------
// Kernel 1: weights -> bf16, transposed for contiguous B-tile loads.
// hlT[h][n][k] = HeadLinear[h][k][n] ; woT[n][k] = OutputLiner[k][n]
// ---------------------------------------------------------------------------
__global__ void convert_weights(const float* __restrict__ HL,
                                const float* __restrict__ WO,
                                bf16_t* __restrict__ hlT,
                                bf16_t* __restrict__ woT) {
  int idx = blockIdx.x * blockDim.x + threadIdx.x;
  int total = Hn * Dn * Dn + En * En;
  for (int i = idx; i < total; i += gridDim.x * blockDim.x) {
    if (i < Hn * Dn * Dn) {
      int h = i / (Dn * Dn), rem = i % (Dn * Dn);
      int n = rem / Dn, k = rem % Dn;
      hlT[i] = (bf16_t)HL[h * Dn * Dn + k * Dn + n];
    } else {
      int j = i - Hn * Dn * Dn;
      int n = j / En, k = j % En;
      woT[j] = (bf16_t)WO[k * En + n];
    }
  }
}

// ---------------------------------------------------------------------------
// Kernel 2: per-head projection. 4 waves/block share one head's 64x64 weight,
// staged once into LDS via async DMA. One wave computes a 16x64 tile (8 WMMA).
// z = 0/1/2 -> Q/K/V.  q,k stored [B,H,L,D]; v stored transposed [B,H,D,L].
// ---------------------------------------------------------------------------
__global__ void __launch_bounds__(128) qkv_proj(
    const float* __restrict__ Q, const float* __restrict__ K,
    const float* __restrict__ V, const bf16_t* __restrict__ hlT,
    bf16_t* __restrict__ qw, bf16_t* __restrict__ kw, bf16_t* __restrict__ vT) {
  __shared__ __align__(32) bf16_t ldsW[64][64];  // 8KB, [n][k]
  int t = threadIdx.x;
  int lane = t & 31, wv = t >> 5;
  int half = lane >> 4, ln = lane & 15;
  int l0 = (blockIdx.x * 4 + wv) * 16;
  int bh = blockIdx.y, b = bh >> 4, h = bh & 15;
  int z = blockIdx.z;

  {  // stage W[h] (8KB contiguous): 128 thr x 64B = 4 x b128 each
    unsigned w0 = lds_off(&ldsW[0][0]);
    const char* gW = (const char*)(hlT + h * Dn * Dn);
#pragma unroll
    for (int i = 0; i < 4; ++i) {
      unsigned off = (unsigned)t * 64u + (unsigned)i * 16u;
      async_b128(w0 + off, gW + off);
    }
  }

  const float* X =
      (z == 0 ? Q : (z == 1 ? K : V)) + ((size_t)b * Ln + l0) * En + h * Dn;
  const float* Xrow = X + (size_t)ln * En;

  v16bf a[2];
#pragma unroll
  for (int ks = 0; ks < 2; ++ks) {
    v8f r0 = *(const v8f*)(Xrow + ks * 32 + half * 16);       // K run 0
    v8f r1 = *(const v8f*)(Xrow + ks * 32 + half * 16 + 8);  // K run 1
#pragma unroll
    for (int e = 0; e < 8; ++e) {
      a[ks][e] = (bf16_t)r0[e];
      a[ks][e + 8] = (bf16_t)r1[e];
    }
  }
  wait_async0();
  __syncthreads();

#pragma unroll
  for (int nt = 0; nt < 4; ++nt) {
    v8f acc = {};
#pragma unroll
    for (int ks = 0; ks < 2; ++ks)
      acc = wmma_bf16(a[ks], ld32(&ldsW[nt * 16 + ln][ks * 32 + half * 16]),
                      acc);
    if (z < 2) {
      bf16_t* O = (z == 0 ? qw : kw) + ((size_t)bh * Ln + l0) * Dn;
#pragma unroll
      for (int r = 0; r < 8; ++r)
        O[(size_t)(r + half * 8) * Dn + nt * 16 + ln] = (bf16_t)acc[r];
    } else {
#pragma unroll
      for (int r = 0; r < 8; ++r)
        vT[((size_t)bh * Dn + nt * 16 + ln) * Ln + l0 + r + half * 8] =
            (bf16_t)acc[r];
    }
  }
}

// ---------------------------------------------------------------------------
// Kernel 3: flash attention. 8 waves/block share one (b,h); each wave owns a
// 16-query tile. Double-buffered async DMA prefetch of the next 32-key K/V
// chunk (4KB+4KB) overlaps WMMA on the current chunk. Online softmax in f32.
// ---------------------------------------------------------------------------
__global__ void __launch_bounds__(256) flash_attn(
    const bf16_t* __restrict__ qw, const bf16_t* __restrict__ kw,
    const bf16_t* __restrict__ vT, const unsigned char* __restrict__ QMask,
    const unsigned char* __restrict__ KMask, bf16_t* __restrict__ Obuf) {
  __shared__ __align__(32) bf16_t ldsK[2][32][64];    // [buf][key][d]  8KB
  __shared__ __align__(32) bf16_t ldsV[2][64][32];    // [buf][d][key]  8KB
  __shared__ __align__(32) bf16_t ldsP[8][16][40];    // per-wave P    10KB
  int t = threadIdx.x;
  int lane = t & 31, wv = t >> 5;
  int half = lane >> 4, ln = lane & 15;
  int q0 = (blockIdx.x * 8 + wv) * 16;
  int bh = blockIdx.y, b = bh >> 4, h = bh & 15;
  const bf16_t* qb = qw + ((size_t)bh * Ln + q0) * Dn;
  const bf16_t* kb = kw + (size_t)bh * Ln * Dn;
  const bf16_t* vb = vT + (size_t)bh * Dn * Ln;
  unsigned lK0 = lds_off(&ldsK[0][0][0]);
  unsigned lV0 = lds_off(&ldsV[0][0][0]);

  // stage one 32-key chunk into buffer `buf` (256 thr: 16B K + 16B V each)
  auto stage = [&](int kc, int buf) {
    unsigned koff = (unsigned)t * 16u;  // K chunk: 4KB contiguous
    async_b128(lK0 + (unsigned)buf * 4096u + koff,
               (const char*)(kb + (size_t)kc * 32 * Dn) + koff);
    int d = t >> 2, q16 = (t & 3) * 16;  // V chunk: 64 rows x 64B
    async_b128(lV0 + (unsigned)buf * 4096u + (unsigned)(d * 64 + q16),
               (const char*)(vb + (size_t)d * Ln + (size_t)kc * 32) + q16);
  };

  v16bf qa[2];
#pragma unroll
  for (int ks = 0; ks < 2; ++ks)
    qa[ks] = ldpair(qb + (size_t)ln * Dn + ks * 32 + half * 16,
                    qb + (size_t)ln * Dn + ks * 32 + half * 16 + 8);

  bool rowMasked[8];  // KMask broadcasts over the key axis => masks query rows
#pragma unroll
  for (int r = 0; r < 8; ++r)
    rowMasked[r] = KMask[b * Ln + q0 + r + half * 8] != 0;

  v8f acc[4] = {};
  float mrow[8], lrow[8];
#pragma unroll
  for (int r = 0; r < 8; ++r) { mrow[r] = -3.0e38f; lrow[r] = 0.0f; }

  stage(0, 0);
  wait_async0();
  __syncthreads();

  constexpr int NC = Ln / 32;
  for (int kc = 0; kc < NC; ++kc) {
    int cur = kc & 1;
    if (kc + 1 < NC) stage(kc + 1, cur ^ 1);  // prefetch overlaps WMMA below

    // --- scores: two 16x16 tiles over 32 keys, K-dim = D = 64 (2 WMMA each)
    v8f s[2];
#pragma unroll
    for (int kt = 0; kt < 2; ++kt) {
      v8f ss = {};
#pragma unroll
      for (int ks = 0; ks < 2; ++ks)
        ss = wmma_bf16(
            qa[ks], ld32(&ldsK[cur][kt * 16 + ln][ks * 32 + half * 16]), ss);
      s[kt] = ss;
    }
    // --- scale 1/sqrt(64) + row mask (-1e9)
#pragma unroll
    for (int kt = 0; kt < 2; ++kt)
#pragma unroll
      for (int r = 0; r < 8; ++r)
        s[kt][r] = rowMasked[r] ? -1.0e9f : s[kt][r] * 0.125f;

    // --- online softmax (per-row; row = vgpr r + 8*half lives in one half)
    float p0[8], p1[8];
#pragma unroll
    for (int r = 0; r < 8; ++r) {
      float tmax = rowmax16(fmaxf(s[0][r], s[1][r]));
      float mnew = fmaxf(mrow[r], tmax);
      float corr = __expf(mrow[r] - mnew);
      p0[r] = __expf(s[0][r] - mnew);
      p1[r] = __expf(s[1][r] - mnew);
      lrow[r] = lrow[r] * corr + rowsum16(p0[r] + p1[r]);
      mrow[r] = mnew;
#pragma unroll
      for (int nt = 0; nt < 4; ++nt) acc[nt][r] *= corr;
    }

    // --- QMask zeroes key columns AFTER the row-sum (reference semantics)
    bool qm0 = QMask[b * Ln + kc * 32 + ln] != 0;
    bool qm1 = QMask[b * Ln + kc * 32 + 16 + ln] != 0;
#pragma unroll
    for (int r = 0; r < 8; ++r) {
      if (qm0) p0[r] = 0.0f;
      if (qm1) p1[r] = 0.0f;
    }

    // --- relayout P: C-layout regs -> LDS row-major -> A-layout bf16
#pragma unroll
    for (int r = 0; r < 8; ++r) {
      ldsP[wv][r + half * 8][ln] = (bf16_t)p0[r];
      ldsP[wv][r + half * 8][16 + ln] = (bf16_t)p1[r];
    }
    asm volatile("s_wait_dscnt 0x0" ::: "memory");  // per-wave LDS RAW fence
    v16bf pa = ldpair(&ldsP[wv][ln][half * 16], &ldsP[wv][ln][half * 16 + 8]);

    // --- P @ V : K = 32 keys, 4 N-tiles of d
#pragma unroll
    for (int nt = 0; nt < 4; ++nt)
      acc[nt] =
          wmma_bf16(pa, ld32(&ldsV[cur][nt * 16 + ln][half * 16]), acc[nt]);

    wait_async0();    // own prefetch landed
    __syncthreads();  // everyone's prefetch landed; cur buffer fully consumed
  }

  // --- finalize: divide by row sum, store bf16 into [B, L, E] (head-packed)
  float inv[8];
#pragma unroll
  for (int r = 0; r < 8; ++r) inv[r] = 1.0f / lrow[r];
#pragma unroll
  for (int nt = 0; nt < 4; ++nt)
#pragma unroll
    for (int r = 0; r < 8; ++r)
      Obuf[((size_t)b * Ln + q0 + r + half * 8) * En + h * Dn + nt * 16 + ln] =
          (bf16_t)(acc[nt][r] * inv[r]);
}

// ---------------------------------------------------------------------------
// Kernel 4: output projection [4096,1024] x [1024,1024]. 4 waves/block share
// the 64-col B panel: each 32-k step's 4KB panel is double-buffer prefetched
// into LDS via async DMA, overlapping 128 WMMA per wave.
// ---------------------------------------------------------------------------
__global__ void __launch_bounds__(128) out_proj(const bf16_t* __restrict__ Obuf,
                                                const bf16_t* __restrict__ woT,
                                                float* __restrict__ Y) {
  __shared__ __align__(32) bf16_t ldsB[2][64][32];  // [buf][n][k] 8KB
  int t = threadIdx.x;
  int lane = t & 31, wv = t >> 5;
  int half = lane >> 4, ln = lane & 15;
  int m0 = (blockIdx.x * 4 + wv) * 16;
  int n0 = blockIdx.y * 64;
  const bf16_t* Arow = Obuf + (size_t)(m0 + ln) * En;
  unsigned lB0 = lds_off(&ldsB[0][0][0]);

  // stage B panel [64n x 32k] for k-step `ks`: 128 thr x 32B (2 x b128)
  auto stageB = [&](int ks, int buf) {
    int n = t >> 1, h32 = (t & 1) * 32;  // 32B half of a 64B row
#pragma unroll
    for (int i = 0; i < 2; ++i) {
      unsigned boff = (unsigned)(h32 + i * 16);
      async_b128(lB0 + (unsigned)buf * 4096u + (unsigned)(n * 64) + boff,
                 (const char*)(woT + (size_t)(n0 + n) * En + ks * 32) + boff);
    }
  };

  v8f acc[4] = {};
  stageB(0, 0);
  wait_async0();
  __syncthreads();

  constexpr int NK = En / 32;
  for (int ks = 0; ks < NK; ++ks) {
    int cur = ks & 1;
    if (ks + 1 < NK) stageB(ks + 1, cur ^ 1);  // prefetch overlaps WMMA

    v16bf a = ldpair(Arow + ks * 32 + half * 16, Arow + ks * 32 + half * 16 + 8);
#pragma unroll
    for (int nt = 0; nt < 4; ++nt)
      acc[nt] =
          wmma_bf16(a, ld32(&ldsB[cur][nt * 16 + ln][half * 16]), acc[nt]);

    wait_async0();
    __syncthreads();
  }
#pragma unroll
  for (int nt = 0; nt < 4; ++nt)
#pragma unroll
    for (int r = 0; r < 8; ++r)
      Y[(size_t)(m0 + r + half * 8) * En + n0 + nt * 16 + ln] = acc[nt][r];
}

// ---------------------------------------------------------------------------
extern "C" void kernel_launch(void* const* d_in, const int* in_sizes, int n_in,
                              void* d_out, int out_size, void* d_ws,
                              size_t ws_size, hipStream_t stream) {
  const float* Q = (const float*)d_in[0];
  const float* K = (const float*)d_in[1];
  const float* V = (const float*)d_in[2];
  const unsigned char* QMask = (const unsigned char*)d_in[3];  // jnp.bool_
  const unsigned char* KMask = (const unsigned char*)d_in[4];
  const float* HL = (const float*)d_in[5];
  const float* WO = (const float*)d_in[6];
  float* Y = (float*)d_out;

  // workspace carve-up (bf16): ~34 MB total
  bf16_t* ws = (bf16_t*)d_ws;
  size_t qkv_elems = (size_t)Bn * Hn * Ln * Dn;    // 4 Mi elems each
  bf16_t* qw = ws;
  bf16_t* kw = qw + qkv_elems;
  bf16_t* vT = kw + qkv_elems;                     // V transposed [B,H,D,L]
  bf16_t* Ob = vT + qkv_elems;                     // [B,L,E]
  bf16_t* hlT = Ob + (size_t)Bn * Ln * En;         // [H,Dout,Din]
  bf16_t* woT = hlT + (size_t)Hn * Dn * Dn;        // [Eout,Ein]

  convert_weights<<<512, 256, 0, stream>>>(HL, WO, hlT, woT);
  qkv_proj<<<dim3(Ln / 64, Bn * Hn, 3), 128, 0, stream>>>(Q, K, V, hlT, qw,
                                                          kw, vT);
  flash_attn<<<dim3(Ln / 128, Bn * Hn), 256, 0, stream>>>(qw, kw, vT, QMask,
                                                          KMask, Ob);
  out_proj<<<dim3((Bn * Ln) / 64, En / 64), 128, 0, stream>>>(Ob, woT, Y);
}
